// Resonator_80427557585392
// MI455X (gfx1250) — compile-verified, matrix operand order checked by reading, not run on hardware
//
#include <hip/hip_runtime.h>
#include <hip/hip_bf16.h>
#include <stdint.h>

#define B_ 512
#define F_ 4
#define M_ 128
#define D_ 8192
#define ITERS_ 16

typedef __attribute__((ext_vector_type(16))) __bf16 bf16x16;
typedef __attribute__((ext_vector_type(8)))  __bf16 bf16x8;
typedef __attribute__((ext_vector_type(8)))  float  f32x8;

// Load a 16x32 bf16 fragment (row-major, leading dim ld) in the CDNA5
// 16-bit A/B per-lane layout: lane%16 = row, lanes<16 hold K {0..7,16..23},
// lanes>=16 hold K {8..15,24..31}; two contiguous 16B chunks per lane.
static __device__ __forceinline__ bf16x16 frag_bf16(const __bf16* p, int ld, int lane) {
  int r = lane & 15;
  int o = (lane & 16) ? 8 : 0;
  const __bf16* row = p + (size_t)r * ld;
  bf16x8 lo = *(const bf16x8*)(row + o);
  bf16x8 hi = *(const bf16x8*)(row + o + 16);
  bf16x16 v;
#pragma unroll
  for (int i = 0; i < 8; ++i) { v[i] = lo[i]; v[i + 8] = hi[i]; }
  return v;
}

// Same fragment layout but source is f32 (the sim matrix), converted to bf16.
static __device__ __forceinline__ bf16x16 frag_f32(const float* p, int ld, int lane) {
  int r = lane & 15;
  int o = (lane & 16) ? 8 : 0;
  const float* row = p + (size_t)r * ld;
  bf16x16 v;
#pragma unroll
  for (int i = 0; i < 8; ++i) { v[i] = (__bf16)row[o + i]; v[i + 8] = (__bf16)row[o + 16 + i]; }
  return v;
}

// ---------------- one-time prep kernels ----------------

__global__ __launch_bounds__(256) void k_cvt_input(const float* __restrict__ in,
                                                   __bf16* __restrict__ out,
                                                   unsigned* __restrict__ diff) {
  size_t i = (size_t)blockIdx.x * 256 + threadIdx.x;
  if (i < ITERS_) diff[i] = 0u;
  size_t base = i * 8;
  if (base >= (size_t)B_ * D_) return;
#pragma unroll
  for (int j = 0; j < 8; ++j) out[base + j] = (__bf16)in[base + j];
}

__global__ __launch_bounds__(256) void k_cvt_cb(const float* __restrict__ cb,
                                                __bf16* __restrict__ cbb,   // [F][M][D]
                                                __bf16* __restrict__ cbt) { // [F][D][M]
  size_t i = (size_t)blockIdx.x * 256 + threadIdx.x;
  size_t base = i * 8;
  if (base >= (size_t)F_ * M_ * D_) return;
  size_t f = base / ((size_t)M_ * D_);
  size_t r = base % ((size_t)M_ * D_);
  size_t m = r / D_;
  size_t d = r % D_;
#pragma unroll
  for (int j = 0; j < 8; ++j) {
    __bf16 v = (__bf16)cb[base + j];
    cbb[base + j] = v;
    cbt[(f * D_ + d + j) * M_ + m] = v;
  }
}

__global__ __launch_bounds__(256) void k_sum_cb(const __bf16* __restrict__ cbb,
                                                __bf16* __restrict__ S) { // [F][D]
  int i = blockIdx.x * 256 + threadIdx.x;
  if (i >= F_ * D_) return;
  int f = i / D_, d = i % D_;
  float s = 0.f;
  for (int m = 0; m < M_; ++m) s += (float)cbb[((size_t)f * M_ + m) * D_ + d];
  S[i] = (__bf16)s;  // |sum| <= 128, integer: exact in bf16
}

__global__ __launch_bounds__(256) void k_init_est(const __bf16* __restrict__ S,
                                                  __bf16* __restrict__ est) { // [F][B][D]
  size_t i = (size_t)blockIdx.x * 256 + threadIdx.x;
  size_t base = i * 8;
  if (base >= (size_t)F_ * B_ * D_) return;
  size_t f = base / ((size_t)B_ * D_);
  size_t d = base % D_;
  *(bf16x8*)(est + base) = *(const bf16x8*)(S + f * D_ + d);
}

// ---------------- per-iteration kernels ----------------

// newest[f][b][d] = input[b][d] * prod_{f'!=f} est[f'][b][d]
__global__ __launch_bounds__(256) void k_prep(const __bf16* __restrict__ inb,
                                              const __bf16* __restrict__ est,
                                              __bf16* __restrict__ ne) {
  size_t i = (size_t)blockIdx.x * 256 + threadIdx.x;
  size_t base = i * 8;
  if (base >= (size_t)B_ * D_) return;
  const size_t P = (size_t)B_ * D_;
  bf16x8 vin = *(const bf16x8*)(inb + base);
  bf16x8 e0 = *(const bf16x8*)(est + 0 * P + base);
  bf16x8 e1 = *(const bf16x8*)(est + 1 * P + base);
  bf16x8 e2 = *(const bf16x8*)(est + 2 * P + base);
  bf16x8 e3 = *(const bf16x8*)(est + 3 * P + base);
  bf16x8 n0, n1, n2, n3;
#pragma unroll
  for (int j = 0; j < 8; ++j) {
    float x = (float)vin[j];
    float a = (float)e0[j], b = (float)e1[j], c = (float)e2[j], d = (float)e3[j];
    float ab = a * b, cd = c * d;
    n0[j] = (__bf16)(x * b * cd);
    n1[j] = (__bf16)(x * a * cd);
    n2[j] = (__bf16)(x * ab * d);
    n3[j] = (__bf16)(x * ab * c);
  }
  *(bf16x8*)(ne + 0 * P + base) = n0;
  *(bf16x8*)(ne + 1 * P + base) = n1;
  *(bf16x8*)(ne + 2 * P + base) = n2;
  *(bf16x8*)(ne + 3 * P + base) = n3;
}

// sim[f][b][m] = sum_d A[f][b][d] * cb[f][m][d]
// grid = F*32*8 WGs; each WG owns one 16x16 output tile, 8 waves split K=8192.
__global__ __launch_bounds__(256) void k_sim(const __bf16* __restrict__ A,
                                             const __bf16* __restrict__ cbb,
                                             float* __restrict__ sim) {
  __shared__ float red[8 * 256];
  int tile = blockIdx.x;
  int mt = tile & 7;  tile >>= 3;
  int bt = tile & 31; tile >>= 5;
  int f  = tile;
  int wave = threadIdx.x >> 5, lane = threadIdx.x & 31;
  const __bf16* Ap = A   + ((size_t)f * B_ + bt * 16) * (size_t)D_;
  const __bf16* Bp = cbb + ((size_t)f * M_ + mt * 16) * (size_t)D_;
  f32x8 acc = {};
  int k0 = wave * (D_ / 8);
#pragma unroll 4
  for (int k = k0; k < k0 + D_ / 8; k += 32) {
    bf16x16 a = frag_bf16(Ap + k, D_, lane);
    bf16x16 b = frag_bf16(Bp + k, D_, lane);
    acc = __builtin_amdgcn_wmma_f32_16x16x32_bf16(false, a, false, b, (short)0, acc, false, false);
  }
#pragma unroll
  for (int v = 0; v < 8; ++v) red[wave * 256 + v * 32 + lane] = acc[v];
  __syncthreads();
  int t = threadIdx.x;
  float s = 0.f;
#pragma unroll
  for (int w = 0; w < 8; ++w) s += red[w * 256 + t];
  int v = t >> 5, l = t & 31;
  int Mr = v + ((l & 16) ? 8 : 0);
  int Nc = l & 15;
  sim[((size_t)f * B_ + (bt * 16 + Mr)) * M_ + (mt * 16 + Nc)] = s;
}

// out[f][b][d] = sum_m sim[f][b][m] * cb[f][m][d]; est = sign(out); diff count.
// grid = F*32*64 WGs; wave w handles d-tile (dg*8+w).
__global__ __launch_bounds__(256) void k_out(const float* __restrict__ sim,
                                             const __bf16* __restrict__ cbt,
                                             __bf16* __restrict__ est,
                                             unsigned* __restrict__ diff, int iter) {
  __shared__ unsigned wg_cnt;
  if (threadIdx.x == 0) wg_cnt = 0u;
  __syncthreads();
  int tile = blockIdx.x;
  int dg = tile & 63; tile >>= 6;
  int bt = tile & 31; tile >>= 5;
  int f  = tile;
  int wave = threadIdx.x >> 5, lane = threadIdx.x & 31;
  int d0 = (dg * 8 + wave) * 16, b0 = bt * 16;
  const float*  Ap = sim + ((size_t)f * B_ + b0) * M_;
  const __bf16* Bp = cbt + ((size_t)f * D_ + d0) * M_;
  f32x8 acc = {};
#pragma unroll
  for (int k = 0; k < M_; k += 32) {
    bf16x16 a = frag_f32(Ap + k, M_, lane);
    bf16x16 b = frag_bf16(Bp + k, M_, lane);
    acc = __builtin_amdgcn_wmma_f32_16x16x32_bf16(false, a, false, b, (short)0, acc, false, false);
  }
  unsigned cnt = 0;
  int n = lane & 15;
  int roff = (lane & 16) ? 8 : 0;
#pragma unroll
  for (int v = 0; v < 8; ++v) {
    float o = acc[v];
    float sv = (o > 0.f) ? 1.f : ((o < 0.f) ? -1.f : 0.f);
    size_t idx = ((size_t)f * B_ + (b0 + v + roff)) * (size_t)D_ + (d0 + n);
    float old = (float)est[idx];
    if (old != sv) ++cnt;
    est[idx] = (__bf16)sv;
  }
  if (cnt) atomicAdd(&wg_cnt, cnt);
  __syncthreads();
  if (threadIdx.x == 0 && wg_cnt) atomicAdd(&diff[iter], wg_cnt);
}

// ---------------- epilogue kernels ----------------

__global__ __launch_bounds__(256) void k_argmax(const float* __restrict__ sim,
                                                int* __restrict__ win) {
  int t = blockIdx.x * 256 + threadIdx.x;
  if (t >= F_ * B_) return;
  int f = t / B_, b = t % B_;
  const float* p = sim + ((size_t)f * B_ + b) * M_;
  float best = p[0]; int bi = 0;
  for (int m = 1; m < M_; ++m) { float v = p[m]; if (v > best) { best = v; bi = m; } }
  win[b * F_ + f] = bi;  // first-max, like jnp.argmax
}

__global__ __launch_bounds__(256) void k_gather(const float* __restrict__ cb,
                                                const int* __restrict__ win,
                                                float* __restrict__ out) {
  size_t i = (size_t)blockIdx.x * 256 + threadIdx.x;
  size_t base = i * 8;
  if (base >= (size_t)B_ * F_ * D_) return;
  size_t b = base / ((size_t)F_ * D_);
  size_t r = base % ((size_t)F_ * D_);
  size_t f = r / D_;
  size_t d = r % D_;
  int m = win[b * F_ + f];
  const float* src = cb + (((size_t)f * M_ + m) * (size_t)D_ + d);
#pragma unroll
  for (int j = 0; j < 8; ++j) out[base + j] = src[j];
}

__global__ void k_final(const unsigned* __restrict__ diff, float* __restrict__ out) {
  if (threadIdx.x == 0 && blockIdx.x == 0) {
    int k = ITERS_ - 1;
    for (int i = 0; i < ITERS_; ++i) if (diff[i] == 0u) { k = i; break; }
    out[(size_t)B_ * F_ * D_] = (float)k;
  }
}

// ---------------- host launcher ----------------

extern "C" void kernel_launch(void* const* d_in, const int* in_sizes, int n_in,
                              void* d_out, int out_size, void* d_ws, size_t ws_size,
                              hipStream_t stream) {
  (void)in_sizes; (void)n_in; (void)out_size; (void)ws_size;
  const float* inp = (const float*)d_in[0];
  const float* cb  = (const float*)d_in[1];
  float* out = (float*)d_out;

  char* ws = (char*)d_ws;
  size_t o = 0;
  auto alloc = [&](size_t bytes) -> char* {
    char* p = ws + o;
    o = (o + bytes + 255) & ~(size_t)255;
    return p;
  };
  __bf16* inb   = (__bf16*)alloc((size_t)B_ * D_ * 2);        //  8.4 MB
  __bf16* cbb   = (__bf16*)alloc((size_t)F_ * M_ * D_ * 2);   //  8.4 MB
  __bf16* cbt   = (__bf16*)alloc((size_t)F_ * D_ * M_ * 2);   //  8.4 MB
  __bf16* est   = (__bf16*)alloc((size_t)F_ * B_ * D_ * 2);   // 33.5 MB
  __bf16* ne    = (__bf16*)alloc((size_t)F_ * B_ * D_ * 2);   // 33.5 MB
  float*  sim   = (float*)alloc((size_t)F_ * B_ * M_ * 4);    //  1.0 MB
  __bf16* S     = (__bf16*)alloc((size_t)F_ * D_ * 2);
  int*    win   = (int*)alloc((size_t)B_ * F_ * 4);
  unsigned* diff = (unsigned*)alloc(ITERS_ * 4);

  const int blk = 256;
  k_cvt_input<<<(B_ * D_ / 8 + blk - 1) / blk, blk, 0, stream>>>(inp, inb, diff);
  k_cvt_cb<<<(F_ * M_ * D_ / 8 + blk - 1) / blk, blk, 0, stream>>>(cb, cbb, cbt);
  k_sum_cb<<<(F_ * D_ + blk - 1) / blk, blk, 0, stream>>>(cbb, S);
  k_init_est<<<(F_ * B_ * D_ / 8 + blk - 1) / blk, blk, 0, stream>>>(S, est);

  for (int it = 0; it < ITERS_; ++it) {
    k_prep<<<(B_ * D_ / 8 + blk - 1) / blk, blk, 0, stream>>>(inb, est, ne);
    k_sim<<<F_ * 32 * 8, blk, 0, stream>>>(ne, cbb, sim);
    k_out<<<F_ * 32 * 64, blk, 0, stream>>>(sim, cbt, est, diff, it);
  }

  k_sim<<<F_ * 32 * 8, blk, 0, stream>>>(est, cbb, sim);
  k_argmax<<<(F_ * B_ + blk - 1) / blk, blk, 0, stream>>>(sim, win);
  k_gather<<<((size_t)B_ * F_ * D_ / 8 + blk - 1) / blk, blk, 0, stream>>>(cb, win, out);
  k_final<<<1, 32, 0, stream>>>(diff, out);
}
